// LorentzGraphConvolution_1400159338517
// MI455X (gfx1250) — matrix operand order repaired and checked by reading, not compiled, not run on hardware
//
#include <hip/hip_runtime.h>

typedef __attribute__((ext_vector_type(16))) __bf16 v16bf;
typedef __attribute__((ext_vector_type(8)))  __bf16 v8bf;
typedef __attribute__((ext_vector_type(8)))  float  v8f;

#define N_NODES 12288
#define DDIM    64

__device__ __forceinline__ __bf16 f2bf(float f) {
  unsigned u = __builtin_bit_cast(unsigned, f);
  unsigned r = u + 0x7FFFu + ((u >> 16) & 1u);
  unsigned short s = (unsigned short)(r >> 16);
  return __builtin_bit_cast(__bf16, s);
}

// fast sigmoid: v_exp_f32 + v_rcp_f32 (no IEEE div chain)
__device__ __forceinline__ float fast_sigmoid(float x) {
  return __builtin_amdgcn_rcpf(1.0f + __expf(-x));
}

// -------------------------------------------------------------------------
// lorentz_linear: h = x @ W^T + b ; time = sigmoid(h0)*exp(scale)+1.1 ;
// xn = h[1:]; s = (t^2-1)/clip(|xn|^2,1e-8); out = [t, xn*sqrt(s)]
// One row per 64-thread block.
// -------------------------------------------------------------------------
__global__ void lorentz_linear_kernel(const float* __restrict__ in,
                                      const float* __restrict__ W,
                                      const float* __restrict__ b,
                                      const float* __restrict__ scale_p,
                                      float* __restrict__ out_f32,
                                      __bf16* __restrict__ out_bf16,
                                      int negate_time)
{
  __shared__ float srow[DDIM];
  __shared__ float sred[DDIM];
  __shared__ float s_h0;
  const int n = blockIdx.x;
  const int d = threadIdx.x;

  srow[d] = in[n * DDIM + d];
  __syncthreads();

  float acc = b[d];
  const float* w = W + d * DDIM;
  #pragma unroll 8
  for (int j = 0; j < DDIM; ++j) acc = fmaf(srow[j], w[j], acc);

  sred[d] = (d == 0) ? 0.0f : acc * acc;
  if (d == 0) s_h0 = acc;
  __syncthreads();
  for (int off = DDIM / 2; off > 0; off >>= 1) {
    if (d < off) sred[d] += sred[d + off];
    __syncthreads();
  }
  const float sq = fmaxf(sred[0], 1e-8f);
  const float t  = fast_sigmoid(s_h0) * __expf(*scale_p) + 1.1f;
  const float sr = sqrtf((t * t - 1.0f) * __builtin_amdgcn_rcpf(sq));
  const float val = (d == 0) ? t : acc * sr;

  if (out_f32)  out_f32[n * DDIM + d] = val;
  if (out_bf16) {
    float v = (negate_time && d == 0) ? -val : val;
    out_bf16[n * DDIM + d] = f2bf(v);
  }
}

// hT[d][n] = bf16(h[n][d]) : coalesced writes, strided (L2-hot) reads
__global__ void transpose_bf16_kernel(const float* __restrict__ h,
                                      __bf16* __restrict__ hT)
{
  const int idx = blockIdx.x * blockDim.x + threadIdx.x; // over DDIM*N
  const int d = idx / N_NODES;
  const int n = idx - d * N_NODES;
  hT[idx] = f2bf(h[n * DDIM + d]);
}

// -------------------------------------------------------------------------
// Fused:  att = adj * sigmoid((2 + 2*qm@k^T)/att_scale + att_bias)
//         support = att @ h ;  out = support / sqrt(clip(|inner|,1e-8))
// One wave per 16 output rows; 4 waves (64 rows) per block.
// -------------------------------------------------------------------------
__global__ __launch_bounds__(128)
void lgc_main_kernel(const __bf16* __restrict__ qm,   // N x 64, time negated
                     const __bf16* __restrict__ kb,   // N x 64
                     const __bf16* __restrict__ hT,   // 64 x N
                     const float*  __restrict__ adj,  // N x N
                     const float*  __restrict__ att_bias_p,
                     const float*  __restrict__ att_scale_p,
                     float* __restrict__ out)         // N x 64
{
  __shared__ alignas(32) __bf16 satt[4][16][32];

  const int wave  = threadIdx.x >> 5;
  const int lane  = threadIdx.x & 31;
  const int lhalf = lane >> 4;      // 0 | 1
  const int l16   = lane & 15;
  const int rowbase = blockIdx.x * 64 + wave * 16;

  // sigmoid argument: (2 + 2c)/s + b  ==  c*(2/s) + (2/s + b)  -> single FMA
  const float two_over_s = 2.0f * __builtin_amdgcn_rcpf(*att_scale_p);
  const float aff_b      = two_over_s + (*att_bias_p);

  union F { v16bf v; v8bf h2[2]; };

  // A fragments of qm (16x64 -> two 16x32 frags), fixed across the whole loop.
  // A k order chosen here (same as B, not AMD's table order): lane M=l16; element e -> K = 16*lhalf + e
  F a0, a1;
  {
    const __bf16* base = qm + (rowbase + l16) * DDIM;
    const int klo = lhalf * 16;
    a0.h2[0] = *(const v8bf*)(base + klo);
    a0.h2[1] = *(const v8bf*)(base + 8 + klo);
    a1.h2[0] = *(const v8bf*)(base + 32 + klo);
    a1.h2[1] = *(const v8bf*)(base + 40 + klo);
  }

  v8f sup0 = {}, sup1 = {}, sup2 = {}, sup3 = {};

  for (int jb = 0; jb < N_NODES; jb += 32) {
    // ---- GEMM1 B fragments: B[dk,n] = k[jb + n][d]
    // B k order chosen here (same as A, not AMD's table order): lane N=l16; element e -> K = 16*lhalf + e
    F bk00, bk10, bk01, bk11;  // [kblk][nblk]
    {
      const int koff = lhalf * 16;
      const __bf16* c0p = kb + (jb + l16) * DDIM + koff;
      const __bf16* c1p = kb + (jb + 16 + l16) * DDIM + koff;
      bk00.v = *(const v16bf*)(c0p);
      bk10.v = *(const v16bf*)(c0p + 32);
      bk01.v = *(const v16bf*)(c1p);
      bk11.v = *(const v16bf*)(c1p + 32);
    }

    v8f c0 = {}, c1 = {};
    c0 = __builtin_amdgcn_wmma_f32_16x16x32_bf16(false, a0.v, false, bk00.v, (short)0, c0, false, false);
    c0 = __builtin_amdgcn_wmma_f32_16x16x32_bf16(false, a1.v, false, bk10.v, (short)0, c0, false, false);
    c1 = __builtin_amdgcn_wmma_f32_16x16x32_bf16(false, a0.v, false, bk01.v, (short)0, c1, false, false);
    c1 = __builtin_amdgcn_wmma_f32_16x16x32_bf16(false, a1.v, false, bk11.v, (short)0, c1, false, false);

    // ---- epilogue: sigmoid + adj mask (adj streamed non-temporally),
    //      stage att tile (16x32 bf16) in LDS to re-layout C-frag -> A-frag
    #pragma unroll
    for (int r = 0; r < 8; ++r) {
      const int m = r + lhalf * 8;
      const long aidx = (long)(rowbase + m) * N_NODES + jb + l16;
      const float av0 = __builtin_nontemporal_load(adj + aidx);
      const float av1 = __builtin_nontemporal_load(adj + aidx + 16);
      const float s0 = fast_sigmoid(fmaf(c0[r], two_over_s, aff_b));
      const float s1 = fast_sigmoid(fmaf(c1[r], two_over_s, aff_b));
      satt[wave][m][l16]      = f2bf(av0 * s0);
      satt[wave][m][16 + l16] = f2bf(av1 * s1);
    }

    // ---- A fragment of att (16x32) from LDS (cross-lane transpose via DS)
    F aatt;
    {
      const __bf16* p = &satt[wave][l16][0];
      const int klo = lhalf * 16;
      aatt.h2[0] = *(const v8bf*)(p + klo);
      aatt.h2[1] = *(const v8bf*)(p + 8 + klo);
    }

    // ---- GEMM2 B fragments: B[kh,n] = h[jb+kh][nblk*16 + l16] = hT[col][jb+kh]
    F bh0, bh1, bh2, bh3;
    {
      const int koff = lhalf * 16;
      bh0.v = *(const v16bf*)(hT + (long)(0  + l16) * N_NODES + jb + koff);
      bh1.v = *(const v16bf*)(hT + (long)(16 + l16) * N_NODES + jb + koff);
      bh2.v = *(const v16bf*)(hT + (long)(32 + l16) * N_NODES + jb + koff);
      bh3.v = *(const v16bf*)(hT + (long)(48 + l16) * N_NODES + jb + koff);
    }

    sup0 = __builtin_amdgcn_wmma_f32_16x16x32_bf16(false, aatt.v, false, bh0.v, (short)0, sup0, false, false);
    sup1 = __builtin_amdgcn_wmma_f32_16x16x32_bf16(false, aatt.v, false, bh1.v, (short)0, sup1, false, false);
    sup2 = __builtin_amdgcn_wmma_f32_16x16x32_bf16(false, aatt.v, false, bh2.v, (short)0, sup2, false, false);
    sup3 = __builtin_amdgcn_wmma_f32_16x16x32_bf16(false, aatt.v, false, bh3.v, (short)0, sup3, false, false);
  }

  // ---- Lorentz row normalization: inner = -s0^2 + sum_{d>0} s_d^2
  // C layout: VGPR r, lane -> row M = r + 8*lhalf, col d = n*16 + l16.
  #pragma unroll
  for (int r = 0; r < 8; ++r) {
    const float v0 = sup0[r], v1 = sup1[r], v2 = sup2[r], v3 = sup3[r];
    float p = v0 * v0 + v1 * v1 + v2 * v2 + v3 * v3;
    if (l16 == 0) p -= 2.0f * v0 * v0;   // d==0 enters with minus sign
    #pragma unroll
    for (int m = 1; m < 16; m <<= 1) p += __shfl_xor(p, m, 16);
    const float rd = __builtin_amdgcn_rsqf(fmaxf(fabsf(p), 1e-8f));
    const int row = rowbase + r + lhalf * 8;
    float* o = out + (long)row * DDIM;
    o[l16]      = v0 * rd;
    o[16 + l16] = v1 * rd;
    o[32 + l16] = v2 * rd;
    o[48 + l16] = v3 * rd;
  }
}

// -------------------------------------------------------------------------
extern "C" void kernel_launch(void* const* d_in, const int* in_sizes, int n_in,
                              void* d_out, int out_size, void* d_ws, size_t ws_size,
                              hipStream_t stream) {
  const float* x         = (const float*)d_in[0];
  const float* adj       = (const float*)d_in[1];
  const float* W         = (const float*)d_in[2];
  const float* b         = (const float*)d_in[3];
  const float* scale     = (const float*)d_in[4];
  const float* Wq        = (const float*)d_in[5];
  const float* bq        = (const float*)d_in[6];
  const float* scale_q   = (const float*)d_in[7];
  const float* Wk        = (const float*)d_in[8];
  const float* bk        = (const float*)d_in[9];
  const float* scale_k   = (const float*)d_in[10];
  const float* att_bias  = (const float*)d_in[11];
  const float* att_scale = (const float*)d_in[12];
  float* out = (float*)d_out;

  char* ws = (char*)d_ws;
  float*  h_f32 = (float*)(ws);                                   // N*64*4 = 3145728 B
  __bf16* qm_bf = (__bf16*)(ws + (size_t)3145728);                // N*64*2 = 1572864 B
  __bf16* k_bf  = (__bf16*)(ws + (size_t)3145728 + 1572864);      // 1572864 B
  __bf16* hT_bf = (__bf16*)(ws + (size_t)3145728 + 2 * 1572864);  // 1572864 B

  // h = lorentz_linear(x, W, b, scale)
  lorentz_linear_kernel<<<N_NODES, DDIM, 0, stream>>>(x, W, b, scale, h_f32, nullptr, 0);
  // qm = lorentz_linear(h, Wq) with time negated, bf16
  lorentz_linear_kernel<<<N_NODES, DDIM, 0, stream>>>(h_f32, Wq, bq, scale_q, nullptr, qm_bf, 1);
  // k = lorentz_linear(h, Wk), bf16
  lorentz_linear_kernel<<<N_NODES, DDIM, 0, stream>>>(h_f32, Wk, bk, scale_k, nullptr, k_bf, 0);
  // hT bf16 for GEMM2 B fragments
  transpose_bf16_kernel<<<(DDIM * N_NODES) / 256, 256, 0, stream>>>(h_f32, hT_bf);
  // fused attention + aggregation + normalization
  lgc_main_kernel<<<N_NODES / 64, 128, 0, stream>>>(qm_bf, k_bf, hT_bf, adj,
                                                    att_bias, att_scale, out);
}
